// FusedWanBlock_85676007620970
// MI455X (gfx1250) — matrix operand order repaired and checked by reading, not compiled, not run on hardware
//
#include <hip/hip_runtime.h>
#include <hip/hip_bf16.h>

// ---------------------------------------------------------------------------
// Problem constants (match reference)
// ---------------------------------------------------------------------------
#define B_   2
#define L_   2048
#define C_   2048
#define H_   16
#define HD_  128
#define F_   8192
#define EPS_ 1e-6f
#define ATTN_SCALE 0.08838834764831845f   // 1/sqrt(128)

typedef unsigned short u16;
typedef __attribute__((ext_vector_type(16))) __bf16 v16bf;
typedef __attribute__((ext_vector_type(8)))  float  v8f;

union Frag16 {            // one WMMA 16-bit operand: 16 bf16 = 32 bytes = 8 VGPRs
    v16bf v;
    uint4 q[2];
};

__device__ __forceinline__ u16 f2bf(float f) {
    unsigned int u = __float_as_uint(f);
    unsigned int r = u + 0x7FFFu + ((u >> 16) & 1u);   // round-nearest-even
    return (u16)(r >> 16);
}
__device__ __forceinline__ v8f v8f_zero() {
    v8f z = {0.f, 0.f, 0.f, 0.f, 0.f, 0.f, 0.f, 0.f};
    return z;
}
__device__ __forceinline__ v8f wmma_bf16(const Frag16& a, const Frag16& b, v8f c) {
    return __builtin_amdgcn_wmma_f32_16x16x32_bf16(
        /*neg_a=*/false, a.v, /*neg_b=*/false, b.v,
        /*c_mod=*/(short)0, c, /*reuse_a=*/false, /*reuse_b=*/false);
}

// ---------------------------------------------------------------------------
// DPP16 row reductions (16-lane butterfly: xor1, xor2, half-mirror, mirror).
// ---------------------------------------------------------------------------
template<int CTRL>
__device__ __forceinline__ float dpp_perm_f(float x) {
    int i = __builtin_bit_cast(int, x);
    int y = __builtin_amdgcn_update_dpp(i, i, CTRL, 0xF, 0xF, true);
    return __builtin_bit_cast(float, y);
}
__device__ __forceinline__ float row_max16(float x) {
    x = fmaxf(x, dpp_perm_f<0xB1>(x));    // quad_perm(1,0,3,2)  : xor 1
    x = fmaxf(x, dpp_perm_f<0x4E>(x));    // quad_perm(2,3,0,1)  : xor 2
    x = fmaxf(x, dpp_perm_f<0x141>(x));   // row_half_mirror     : other quad of 8
    x = fmaxf(x, dpp_perm_f<0x140>(x));   // row_mirror          : other half of 16
    return x;
}
__device__ __forceinline__ float row_sum16(float x) {
    x += dpp_perm_f<0xB1>(x);
    x += dpp_perm_f<0x4E>(x);
    x += dpp_perm_f<0x141>(x);
    x += dpp_perm_f<0x140>(x);
    return x;
}

// ---------------------------------------------------------------------------
// CDNA5 async global -> LDS DMA (ASYNCcnt-tracked, no VGPR staging)
// ---------------------------------------------------------------------------
#define ASYNC_LOAD_B128(ldsPtr, gPtr)                                     \
    asm volatile("global_load_async_to_lds_b128 %0, %1, off"              \
                 :: "v"((unsigned)(uintptr_t)(ldsPtr)),                   \
                    "v"((const void*)(gPtr))                              \
                 : "memory")
#define WAIT_ASYNCCNT0() asm volatile("s_wait_asynccnt 0x0" ::: "memory")

// ---------------------------------------------------------------------------
// Kernel 1: one-time transpose-cast  W[K][N] fp32 -> Wt[N][K] bf16.
// Tiled through LDS so both global read and write are coalesced.
// ---------------------------------------------------------------------------
__global__ __launch_bounds__(256)
void castT_bf16_kernel(const float* __restrict__ W, u16* __restrict__ Wt,
                       int K, int N) {
    __shared__ __align__(16) u16 tile[64][72];   // [n][k], padded
    const int k0 = blockIdx.y * 64;
    const int n0 = blockIdx.x * 64;
    const int t  = threadIdx.x;
    #pragma unroll
    for (int p = 0; p < 4; ++p) {
        int idx = t * 4 + p * 1024;          // 4 consecutive floats along n
        int kk = idx >> 6, nn = idx & 63;
        float4 v = *(const float4*)(W + (size_t)(k0 + kk) * N + n0 + nn);
        tile[nn + 0][kk] = f2bf(v.x);
        tile[nn + 1][kk] = f2bf(v.y);
        tile[nn + 2][kk] = f2bf(v.z);
        tile[nn + 3][kk] = f2bf(v.w);
    }
    __syncthreads();
    #pragma unroll
    for (int p = 0; p < 2; ++p) {
        int idx = t * 8 + p * 2048;          // 8 consecutive bf16 along k
        int nn = idx >> 6, kk = idx & 63;
        *(uint4*)(Wt + (size_t)(n0 + nn) * K + k0 + kk) = *(const uint4*)&tile[nn][kk];
    }
}

// ---------------------------------------------------------------------------
// Kernel 2: LayerNorm + adaLN modulate -> bf16.  One (b,l) row per block.
// ---------------------------------------------------------------------------
__global__ __launch_bounds__(256)
void ln_mod_kernel(const float* __restrict__ x,
                   const float* __restrict__ lw, const float* __restrict__ lb,
                   const float* __restrict__ scale, const float* __restrict__ shift,
                   u16* __restrict__ out) {
    const int row = blockIdx.x;              // b*L + l
    const int b   = row / L_;
    const float* xp = x + (size_t)row * C_;
    __shared__ float red[16];

    float s = 0.f, ss = 0.f;
    for (int i = threadIdx.x; i < C_; i += 256) {
        float v = xp[i];
        s += v; ss += v * v;
    }
    s  = row_sum16(s);
    ss = row_sum16(ss);
    s  += __shfl_xor(s,  16, 32);
    ss += __shfl_xor(ss, 16, 32);
    const int lane = threadIdx.x & 31, wave = threadIdx.x >> 5;
    if (lane == 0) { red[wave] = s; red[8 + wave] = ss; }
    __syncthreads();
    float S = 0.f, SS = 0.f;
    #pragma unroll
    for (int w = 0; w < 8; ++w) { S += red[w]; SS += red[8 + w]; }
    const float mean = S * (1.f / C_);
    const float var  = SS * (1.f / C_) - mean * mean;
    const float rstd = rsqrtf(var + EPS_);

    const float* scb = scale + (size_t)b * C_;
    const float* shb = shift + (size_t)b * C_;
    u16* op = out + (size_t)row * C_;
    for (int i = threadIdx.x; i < C_; i += 256) {
        float xn = (xp[i] - mean) * rstd * lw[i] + lb[i];
        op[i] = f2bf((1.f + scb[i]) * xn + shb[i]);
    }
}

// ---------------------------------------------------------------------------
// Kernel 3: bf16 WMMA GEMM, D = A(MxK) * Bt(NxK)^T, fp32 accumulate.
//  Workgroup tile 64x64 (8 waves, each 16x32).  K stepped by 32.
//  BOTH A and B tiles staged by GLOBAL_LOAD_ASYNC_TO_LDS_B128 — the inner
//  loop has no VALU-staged LDS stores at all.
//  MODE 0: store fp32
//  MODE 1: outF = resid + gate[b,n] * acc
//  MODE 2: dual-B SwiGLU: outH = bf16( silu(A*B0^T) * (A*B1^T) )
// ---------------------------------------------------------------------------
template<int MODE>
__global__ __launch_bounds__(256)
void gemm_bf16_kernel(const u16* __restrict__ A,
                      const u16* __restrict__ B0t,    // [N][K]
                      const u16* __restrict__ B1t,    // [N][K]
                      float* __restrict__ outF,
                      u16*   __restrict__ outH,
                      const float* __restrict__ resid,
                      const float* __restrict__ gate,
                      int M, int N, int K, int Lrows) {
    __shared__ __align__(16) u16 smA [64][40];   // [m][k]
    __shared__ __align__(16) u16 smB [64][40];   // [n][k]
    __shared__ __align__(16) u16 smB2[64][40];   // [n][k]  (MODE 2)

    const int m0 = blockIdx.y * 64;
    const int n0 = blockIdx.x * 64;
    const int t    = threadIdx.x;
    const int wave = t >> 5;
    const int lane = t & 31;
    const int half = lane >> 4;
    const int r    = lane & 15;
    const int waveM = wave >> 1;     // 0..3
    const int waveN = wave & 1;      // 0..1

    // per-thread tile coordinates for the async copies (8 bf16 = one B128)
    const int idx = t * 8;
    const int ar = idx >> 5, ac = idx & 31;      // A tile: 64 rows x 32 k

    v8f acc0 = v8f_zero(), acc1 = v8f_zero();
    v8f acc2 = v8f_zero(), acc3 = v8f_zero();

    for (int k0 = 0; k0 < K; k0 += 32) {
        // ---- async DMA: A 64x32 and B(t) 64x32 straight into LDS ----
        ASYNC_LOAD_B128(&smA[ar][ac], A   + (size_t)(m0 + ar) * K + k0 + ac);
        ASYNC_LOAD_B128(&smB[ar][ac], B0t + (size_t)(n0 + ar) * K + k0 + ac);
        if (MODE == 2)
            ASYNC_LOAD_B128(&smB2[ar][ac], B1t + (size_t)(n0 + ar) * K + k0 + ac);
        if (k0 + 32 < K) {
            __builtin_prefetch(A   + (size_t)(m0 + ar) * K + k0 + 32 + ac, 0, 1);
            __builtin_prefetch(B0t + (size_t)(n0 + ar) * K + k0 + 32 + ac, 0, 1);
        }
        WAIT_ASYNCCNT0();
        __syncthreads();

        // ---- A fragment: rows waveM*16 + r, contiguous K per half-wave {half*16..} ----
        Frag16 fa;
        fa.q[0] = *(const uint4*)&smA[waveM * 16 + r][half * 16];
        fa.q[1] = *(const uint4*)&smA[waveM * 16 + r][half * 16 + 8];

        // ---- B fragments: one column per lane, contiguous K per half-wave ----
        const int cA = waveN * 32 + r;
        const int cB = cA + 16;
        Frag16 fb0, fb1;
        fb0.q[0] = *(const uint4*)&smB[cA][half * 16];
        fb0.q[1] = *(const uint4*)&smB[cA][half * 16 + 8];
        fb1.q[0] = *(const uint4*)&smB[cB][half * 16];
        fb1.q[1] = *(const uint4*)&smB[cB][half * 16 + 8];

        acc0 = wmma_bf16(fa, fb0, acc0);
        acc1 = wmma_bf16(fa, fb1, acc1);
        if (MODE == 2) {
            Frag16 g0, g1;
            g0.q[0] = *(const uint4*)&smB2[cA][half * 16];
            g0.q[1] = *(const uint4*)&smB2[cA][half * 16 + 8];
            g1.q[0] = *(const uint4*)&smB2[cB][half * 16];
            g1.q[1] = *(const uint4*)&smB2[cB][half * 16 + 8];
            acc2 = wmma_bf16(fa, g0, acc2);
            acc3 = wmma_bf16(fa, g1, acc3);
        }
        __syncthreads();
    }

    // ---- epilogue.  C/D layout: row = half*8+i, col = lane&15 ----
    #pragma unroll
    for (int i = 0; i < 8; ++i) {
        const int m  = m0 + waveM * 16 + half * 8 + i;
        const int nA = n0 + waveN * 32 + r;
        const int nB = nA + 16;
        if (MODE == 0) {
            outF[(size_t)m * N + nA] = acc0[i];
            outF[(size_t)m * N + nB] = acc1[i];
        } else if (MODE == 1) {
            const int b = m / Lrows;
            outF[(size_t)m * N + nA] = resid[(size_t)m * N + nA] + gate[(size_t)b * N + nA] * acc0[i];
            outF[(size_t)m * N + nB] = resid[(size_t)m * N + nB] + gate[(size_t)b * N + nB] * acc1[i];
        } else { // MODE 2: silu(h1) * h2 -> bf16
            float h1a = acc0[i], h2a = acc2[i];
            float h1b = acc1[i], h2b = acc3[i];
            float ya = h1a * (1.f / (1.f + __expf(-h1a))) * h2a;
            float yb = h1b * (1.f / (1.f + __expf(-h1b))) * h2b;
            outH[(size_t)m * N + nA] = f2bf(ya);
            outH[(size_t)m * N + nB] = f2bf(yb);
        }
    }
}

// ---------------------------------------------------------------------------
// Kernel 4: RoPE(q,k) + split qkv.
//   q,k -> [B][H][L][HD] bf16 (row-major per head)
//   v   -> [B][H][HD][L] bf16 (d-major, so flash V tiles DMA straight to LDS)
// ---------------------------------------------------------------------------
__global__ __launch_bounds__(256)
void rope_split_kernel(const float* __restrict__ qkv,     // [B*L][3*C]
                       const float* __restrict__ freqs,   // [L][HD] (cos|sin)
                       u16* __restrict__ qo, u16* __restrict__ ko,
                       u16* __restrict__ vt) {
    const int row = blockIdx.x;        // b*L + l
    const int b = row / L_, l = row % L_;
    const float* base = qkv + (size_t)row * 3 * C_;
    for (int p = threadIdx.x; p < H_ * (HD_ / 2); p += 256) {
        const int h = p >> 6;              // / (HD/2)
        const int j = p & 63;
        const float c  = freqs[(size_t)l * HD_ + j];
        const float sn = freqs[(size_t)l * HD_ + (HD_ / 2) + j];
        const size_t o = (((size_t)b * H_ + h) * L_ + l) * HD_ + 2 * j;
        // q
        float xr = base[h * HD_ + 2 * j], xi = base[h * HD_ + 2 * j + 1];
        qo[o]     = f2bf(xr * c - xi * sn);
        qo[o + 1] = f2bf(xr * sn + xi * c);
        // k
        xr = base[C_ + h * HD_ + 2 * j]; xi = base[C_ + h * HD_ + 2 * j + 1];
        ko[o]     = f2bf(xr * c - xi * sn);
        ko[o + 1] = f2bf(xr * sn + xi * c);
        // v, transposed layout [b][h][d][l]
        const size_t ov = (((size_t)b * H_ + h) * HD_ + 2 * j) * L_ + l;
        vt[ov]      = f2bf(base[2 * C_ + h * HD_ + 2 * j]);
        vt[ov + L_] = f2bf(base[2 * C_ + h * HD_ + 2 * j + 1]);
    }
}

// ---------------------------------------------------------------------------
// Kernel 5: flash attention.  grid (B*H, L/128); 8 waves; wave = 16 q-rows.
//  Q,K bf16 [B][H][L][HD]; V bf16 [B][H][HD][L]; out bf16 [B][L][C].
//  K and V tiles: async DMA to LDS (no transposition needed anywhere).
//  Softmax row reductions: DPP16 (no LDS).
// ---------------------------------------------------------------------------
__global__ __launch_bounds__(256)
void flash_attn_kernel(const u16* __restrict__ Q, const u16* __restrict__ Km,
                       const u16* __restrict__ Vt, u16* __restrict__ O) {
    __shared__ __align__(16) u16 smK[32][136];      // [key][d]
    __shared__ __align__(16) u16 smV[128][40];      // [d][key]
    __shared__ __align__(16) u16 smP[8][16][40];    // per-wave P re-stripe scratch

    const int bh    = blockIdx.x;            // b*H + h
    const int b     = bh / H_, h = bh % H_;
    const int qBase = blockIdx.y * 128;
    const int t     = threadIdx.x;
    const int wave  = t >> 5;
    const int lane  = t & 31;
    const int half  = lane >> 4;
    const int r     = lane & 15;

    // ---- Q fragments (held in registers for the whole loop) ----
    Frag16 qf[4];
    {
        const u16* qrow = Q + (((size_t)bh * L_) + qBase + wave * 16 + r) * HD_;
        #pragma unroll
        for (int f = 0; f < 4; ++f) {
            qf[f].q[0] = *(const uint4*)(qrow + f * 32 + half * 16);
            qf[f].q[1] = *(const uint4*)(qrow + f * 32 + half * 16 + 8);
        }
    }

    v8f acc[8];
    #pragma unroll
    for (int n = 0; n < 8; ++n) acc[n] = v8f_zero();
    float mrow[8], lsum[8];
    #pragma unroll
    for (int i = 0; i < 8; ++i) { mrow[i] = -3.0e38f; lsum[i] = 0.f; }

    for (int j0 = 0; j0 < L_; j0 += 32) {
        // ---- async DMA: K tile (32 keys x 128 d) and V tile (128 d x 32 keys) ----
        #pragma unroll
        for (int u = 0; u < 2; ++u) {
            const int idx = t * 8 + u * 2048;
            const int kk = idx >> 7, dd = idx & 127;        // K tile coords
            ASYNC_LOAD_B128(&smK[kk][dd],
                            Km + (((size_t)bh * L_) + j0 + kk) * HD_ + dd);
            const int dv = idx >> 5, kv = idx & 31;         // V tile coords
            ASYNC_LOAD_B128(&smV[dv][kv],
                            Vt + ((size_t)bh * HD_ + dv) * L_ + j0 + kv);
        }
        WAIT_ASYNCCNT0();
        __syncthreads();

        // ---- scores: two 16x16 tiles over this 32-key block ----
        v8f s0 = v8f_zero(), s1 = v8f_zero();
        #pragma unroll
        for (int f = 0; f < 4; ++f) {
            Frag16 kb0, kb1;
            kb0.q[0] = *(const uint4*)&smK[r][f * 32 + half * 16];
            kb0.q[1] = *(const uint4*)&smK[r][f * 32 + half * 16 + 8];
            kb1.q[0] = *(const uint4*)&smK[16 + r][f * 32 + half * 16];
            kb1.q[1] = *(const uint4*)&smK[16 + r][f * 32 + half * 16 + 8];
            s0 = wmma_bf16(qf[f], kb0, s0);
            s1 = wmma_bf16(qf[f], kb1, s1);
        }

        // ---- online softmax (DPP16 row reductions across the half-wave) ----
        float alpha[8];
        #pragma unroll
        for (int i = 0; i < 8; ++i) {
            float a = s0[i] * ATTN_SCALE;
            float c = s1[i] * ATTN_SCALE;
            float mt = row_max16(fmaxf(a, c));
            const float mn = fmaxf(mrow[i], mt);
            alpha[i] = __expf(mrow[i] - mn);
            mrow[i]  = mn;
            a = __expf(a - mn);
            c = __expf(c - mn);
            s0[i] = a; s1[i] = c;
            lsum[i] = lsum[i] * alpha[i] + row_sum16(a + c);
        }
        #pragma unroll
        for (int n = 0; n < 8; ++n)
            #pragma unroll
            for (int i = 0; i < 8; ++i) acc[n][i] *= alpha[i];

        // ---- re-stripe P (C-layout) into A-fragment layout via LDS ----
        #pragma unroll
        for (int i = 0; i < 8; ++i) {
            smP[wave][half * 8 + i][r]      = f2bf(s0[i]);
            smP[wave][half * 8 + i][16 + r] = f2bf(s1[i]);
        }
        Frag16 pf;
        pf.q[0] = *(const uint4*)&smP[wave][r][half * 16];
        pf.q[1] = *(const uint4*)&smP[wave][r][half * 16 + 8];

        // ---- O += P(16x32) x V(32x128) ----
        #pragma unroll
        for (int n = 0; n < 8; ++n) {
            Frag16 vb;
            vb.q[0] = *(const uint4*)&smV[n * 16 + r][half * 16];
            vb.q[1] = *(const uint4*)&smV[n * 16 + r][half * 16 + 8];
            acc[n] = wmma_bf16(pf, vb, acc[n]);
        }
        __syncthreads();
    }

    // ---- finalize: O /= l, write bf16 row-major [B][L][C] ----
    #pragma unroll
    for (int n = 0; n < 8; ++n) {
        #pragma unroll
        for (int i = 0; i < 8; ++i) {
            const float o = acc[n][i] / lsum[i];
            const size_t m = (size_t)b * L_ + qBase + wave * 16 + half * 8 + i;
            O[m * C_ + h * HD_ + n * 16 + r] = f2bf(o);
        }
    }
}

// ---------------------------------------------------------------------------
// Host launcher
// ---------------------------------------------------------------------------
extern "C" void kernel_launch(void* const* d_in, const int* in_sizes, int n_in,
                              void* d_out, int out_size, void* d_ws, size_t ws_size,
                              hipStream_t stream) {
    const float* x         = (const float*)d_in[0];
    const float* shift_msa = (const float*)d_in[1];
    const float* scale_msa = (const float*)d_in[2];
    const float* gate_msa  = (const float*)d_in[3];
    const float* shift_mlp = (const float*)d_in[4];
    const float* scale_mlp = (const float*)d_in[5];
    const float* gate_mlp  = (const float*)d_in[6];
    const float* freqs     = (const float*)d_in[7];
    const float* Wqkv      = (const float*)d_in[8];
    const float* Wo        = (const float*)d_in[9];
    const float* W1        = (const float*)d_in[10];
    const float* W2        = (const float*)d_in[11];
    const float* W3        = (const float*)d_in[12];
    const float* ln1_w     = (const float*)d_in[13];
    const float* ln1_b     = (const float*)d_in[14];
    const float* ln2_w     = (const float*)d_in[15];
    const float* ln2_b     = (const float*)d_in[16];

    const int BL = B_ * L_;
    char* ws = (char*)d_ws;
    size_t off = 0;
    auto give = [&](size_t bytes) -> char* {
        char* p = ws + off;
        off = (off + bytes + 255) & ~(size_t)255;
        return p;
    };

    u16*   WqkvT  = (u16*)  give((size_t)C_ * 3 * C_ * sizeof(u16));   // [3C][C]
    u16*   WoT    = (u16*)  give((size_t)C_ * C_ * sizeof(u16));       // [C][C]
    u16*   W1T    = (u16*)  give((size_t)C_ * F_ * sizeof(u16));       // [F][C]
    u16*   W2T    = (u16*)  give((size_t)C_ * F_ * sizeof(u16));       // [F][C]
    u16*   W3T    = (u16*)  give((size_t)F_ * C_ * sizeof(u16));       // [C][F]
    u16*   xmod   = (u16*)  give((size_t)BL * C_ * sizeof(u16));
    float* qkv_f  = (float*)give((size_t)BL * 3 * C_ * sizeof(float));
    u16*   qh     = (u16*)  give((size_t)BL * C_ * sizeof(u16));
    u16*   kh     = (u16*)  give((size_t)BL * C_ * sizeof(u16));
    u16*   vt     = (u16*)  give((size_t)BL * C_ * sizeof(u16));
    u16*   attnh  = (u16*)  give((size_t)BL * C_ * sizeof(u16));
    float* x1     = (float*)give((size_t)BL * C_ * sizeof(float));
    u16*   xmod2  = (u16*)  give((size_t)BL * C_ * sizeof(u16));
    u16*   hidden = (u16*)  give((size_t)BL * F_ * sizeof(u16));

    // 1) one-time transpose-casts: W[K][N] fp32 -> Wt[N][K] bf16
    castT_bf16_kernel<<<dim3(3 * C_ / 64, C_ / 64), 256, 0, stream>>>(Wqkv, WqkvT, C_, 3 * C_);
    castT_bf16_kernel<<<dim3(C_ / 64,     C_ / 64), 256, 0, stream>>>(Wo,   WoT,   C_, C_);
    castT_bf16_kernel<<<dim3(F_ / 64,     C_ / 64), 256, 0, stream>>>(W1,   W1T,   C_, F_);
    castT_bf16_kernel<<<dim3(F_ / 64,     C_ / 64), 256, 0, stream>>>(W2,   W2T,   C_, F_);
    castT_bf16_kernel<<<dim3(C_ / 64,     F_ / 64), 256, 0, stream>>>(W3,   W3T,   F_, C_);

    // 2) LN1 + modulate
    ln_mod_kernel<<<BL, 256, 0, stream>>>(x, ln1_w, ln1_b, scale_msa, shift_msa, xmod);

    // 3) QKV GEMM (fp32 out)
    gemm_bf16_kernel<0><<<dim3(3 * C_ / 64, BL / 64), 256, 0, stream>>>(
        xmod, WqkvT, nullptr, qkv_f, nullptr, nullptr, nullptr, BL, 3 * C_, C_, L_);

    // 4) RoPE + head split (V written d-major)
    rope_split_kernel<<<BL, 256, 0, stream>>>(qkv_f, freqs, qh, kh, vt);

    // 5) flash attention
    flash_attn_kernel<<<dim3(B_ * H_, L_ / 128), 256, 0, stream>>>(qh, kh, vt, attnh);

    // 6) Wo GEMM + residual + gate_msa
    gemm_bf16_kernel<1><<<dim3(C_ / 64, BL / 64), 256, 0, stream>>>(
        attnh, WoT, nullptr, x1, nullptr, x, gate_msa, BL, C_, C_, L_);

    // 7) LN2 + modulate
    ln_mod_kernel<<<BL, 256, 0, stream>>>(x1, ln2_w, ln2_b, scale_mlp, shift_mlp, xmod2);

    // 8) MLP up: dual GEMM + SwiGLU -> bf16 hidden
    gemm_bf16_kernel<2><<<dim3(F_ / 64, BL / 64), 256, 0, stream>>>(
        xmod2, W1T, W2T, nullptr, hidden, nullptr, nullptr, BL, F_, C_, L_);

    // 9) MLP down + residual + gate_mlp -> final output (fp32)
    gemm_bf16_kernel<1><<<dim3(C_ / 64, BL / 64), 256, 0, stream>>>(
        hidden, W3T, nullptr, (float*)d_out, nullptr, x1, gate_mlp, BL, C_, F_, L_);
}
